// MultiHeadAttention_82867099009399
// MI455X (gfx1250) — matrix operand order repaired and checked by reading, not compiled, not run on hardware
//
#include <hip/hip_runtime.h>

#define NUM_HEADS 16
#define D_MODEL   1024
#define D_K       64
#define BATCH     2
#define SEQ       2048

typedef _Float16 v16h __attribute__((ext_vector_type(16)));
typedef float    v8f  __attribute__((ext_vector_type(8)));
typedef float    v4f  __attribute__((ext_vector_type(4)));
typedef int      v4i  __attribute__((ext_vector_type(4)));

union H8  { _Float16 h[8];  v4i q;    };
union H16 { _Float16 h[16]; v4i q[2]; v16h v; };

__device__ __forceinline__ v8f wmma_f16(v16h a, v16h b, v8f c) {
  // D = A(16x32 f16) * B(32x16 f16) + C(16x16 f32)
  return __builtin_amdgcn_wmma_f32_16x16x32_f16(false, a, false, b, (short)0, c,
                                                false, false);
}

// Async global -> LDS copy, 16B per lane.  Tracked with ASYNCcnt (separate
// from the compiler-managed LOADcnt/DScnt), waited with s_wait_asynccnt.
// lds_byte_off = low 32 bits of the generic pointer to __shared__ (= LDS addr).
__device__ __forceinline__ void async_copy_b128(unsigned lds_byte_off,
                                                const void* gptr) {
  asm volatile("global_load_async_to_lds_b128 %0, %1, off"
               :: "v"(lds_byte_off), "v"(gptr) : "memory");
}
__device__ __forceinline__ void wait_async0() {
  asm volatile("s_wait_asynccnt 0" ::: "memory");
}

// ---------------------------------------------------------------------------
// Kernel 1: fused QKV projection.  Out = In @ W^T + bias  (f32 -> f16 WMMA)
// block = 256 threads (8 waves), tile 64 rows x 128 cols, k-step 32.
// Q,K stored [b,h,s,d] f16 ; V stored transposed [b,h,d,s] f16.
// ---------------------------------------------------------------------------
#define LDSTR 40  // halves per LDS row (32 + 8 pad) -> 80B, 16B aligned rows

__global__ __launch_bounds__(256) void qkv_proj_kernel(
    const float* __restrict__ x,  const float* __restrict__ y,
    const float* __restrict__ wq, const float* __restrict__ bq,
    const float* __restrict__ wk, const float* __restrict__ bk,
    const float* __restrict__ wv, const float* __restrict__ bv,
    _Float16* __restrict__ Qb, _Float16* __restrict__ Kb,
    _Float16* __restrict__ Vt)
{
  const int matid = blockIdx.z;                       // 0=Q 1=K 2=V
  const float* In   = (matid == 0) ? x  : y;
  const float* W    = (matid == 0) ? wq : (matid == 1) ? wk : wv;
  const float* Bias = (matid == 0) ? bq : (matid == 1) ? bk : bv;

  __shared__ __align__(16) _Float16 ldsA[64  * LDSTR];
  __shared__ __align__(16) _Float16 ldsB[128 * LDSTR];

  const int tid  = threadIdx.x;
  const int lane = tid & 31;
  const int wave = tid >> 5;
  const int wm   = wave & 3;            // 4 row tiles of 16
  const int wn   = wave >> 2;           // 2 col halves of 64
  const int row0 = blockIdx.y * 64;
  const int col0 = blockIdx.x * 128;

  v8f acc[4] = {};

  for (int k0 = 0; k0 < D_MODEL; k0 += 32) {
    __syncthreads();
    {   // stage A: 64 rows x 32 k, f32 -> f16
      const int r = tid >> 2, c = (tid & 3) * 8;
      const float* src = In + (size_t)(row0 + r) * D_MODEL + k0 + c;
      v4f f0 = *(const v4f*)src;
      v4f f1 = *(const v4f*)(src + 4);
      H8 h8;
#pragma unroll
      for (int e = 0; e < 4; ++e) { h8.h[e] = (_Float16)f0[e]; h8.h[4+e] = (_Float16)f1[e]; }
      *(v4i*)&ldsA[r * LDSTR + c] = h8.q;
    }
    {   // stage B^T: 128 cols(j) x 32 k, row j of W is contiguous in k
      const int j = tid >> 1, c = (tid & 1) * 16;
      const float* src = W + (size_t)(col0 + j) * D_MODEL + k0 + c;
      v4f w0 = *(const v4f*)src,      w1 = *(const v4f*)(src + 4);
      v4f w2 = *(const v4f*)(src + 8), w3 = *(const v4f*)(src + 12);
      H8 lo, hi;
#pragma unroll
      for (int e = 0; e < 4; ++e) {
        lo.h[e] = (_Float16)w0[e]; lo.h[4+e] = (_Float16)w1[e];
        hi.h[e] = (_Float16)w2[e]; hi.h[4+e] = (_Float16)w3[e];
      }
      *(v4i*)&ldsB[j * LDSTR + c]     = lo.q;
      *(v4i*)&ldsB[j * LDSTR + c + 8] = hi.q;
    }
    __syncthreads();

    H16 a;   // A layout: lanes 0-15 row=lane K0-15 ; lanes 16-31 K16-31
    {
      const int r  = wm * 16 + (lane & 15);
      const int kb = (lane < 16) ? 0 : 16;
      a.q[0] = *(const v4i*)&ldsA[r * LDSTR + kb];
      a.q[1] = *(const v4i*)&ldsA[r * LDSTR + kb + 8];
    }
#pragma unroll
    for (int nt = 0; nt < 4; ++nt) {   // B layout: lane = column, 16 contig K
      const int j  = wn * 64 + nt * 16 + (lane & 15);
      const int ks = (lane < 16) ? 0 : 16;
      H16 bm;
      bm.q[0] = *(const v4i*)&ldsB[j * LDSTR + ks];
      bm.q[1] = *(const v4i*)&ldsB[j * LDSTR + ks + 8];
      acc[nt] = wmma_f16(a.v, bm.v, acc[nt]);
    }
  }

  // epilogue: bias + scatter into attention layouts
  const int rowbase = row0 + wm * 16 + ((lane < 16) ? 0 : 8);
#pragma unroll
  for (int nt = 0; nt < 4; ++nt) {
    const int j = col0 + wn * 64 + nt * 16 + (lane & 15);
    const int h = j >> 6, d = j & 63;
    const float bias = Bias[j];
#pragma unroll
    for (int i = 0; i < 8; ++i) {
      const int n  = rowbase + i;
      const int bb = n >> 11, s = n & (SEQ - 1);
      const float vv = acc[nt][i] + bias;
      if (matid == 2)
        Vt[((size_t)(bb * NUM_HEADS + h) * D_K + d) * SEQ + s] = (_Float16)vv;
      else if (matid == 0)
        Qb[((size_t)(bb * NUM_HEADS + h) * SEQ + s) * D_K + d] = (_Float16)vv;
      else
        Kb[((size_t)(bb * NUM_HEADS + h) * SEQ + s) * D_K + d] = (_Float16)vv;
    }
  }
}

// ---------------------------------------------------------------------------
// Kernel 2: flash attention.  block = 256 threads (8 waves), each wave owns a
// 16-query tile (128 queries / block).  K/V chunks of 32 keys are staged in
// double-buffered LDS with async global->LDS copies shared by all 8 waves;
// online softmax; P bounced through per-wave LDS for D->A re-layout.
// ---------------------------------------------------------------------------
#define KSTR 72   // ldsK row stride in halves (64 + 8 pad) -> 144B
#define VSTR 40   // ldsV row stride in halves (32 + 8 pad) -> 80B

__global__ __launch_bounds__(256) void attn_kernel(
    const _Float16* __restrict__ Qb, const _Float16* __restrict__ Kb,
    const _Float16* __restrict__ Vt, const float* __restrict__ mask,
    _Float16* __restrict__ Zb)
{
  __shared__ __align__(16) _Float16 ldsK[2][32 * KSTR]; // 32 keys x 64 d
  __shared__ __align__(16) _Float16 ldsV[2][64 * VSTR]; // 64 d x 32 keys
  __shared__ __align__(16) _Float16 ldsP[8][16 * 32];

  const int tid  = threadIdx.x;
  const int lane = tid & 31;
  const int wave = tid >> 5;
  const int bh   = blockIdx.y;          // b*16 + h
  const int b    = bh >> 4;
  const int q0   = blockIdx.x * 128 + wave * 16;
  const int rr   = lane & 15;
  const int kb   = (lane < 16) ? 0 : 16;  // A-layout k base
  const int ks   = (lane < 16) ? 0 : 16;  // B-layout k base

  // per-thread staging addresses (K: 8 x 16B chunks/row, V: 4 chunks/row)
  const _Float16* Kg_base =
      Kb + (size_t)bh * SEQ * D_K + (size_t)(tid >> 3) * D_K + (tid & 7) * 8;
  const _Float16* Vg_base =
      Vt + ((size_t)bh * D_K + (tid >> 2)) * SEQ + (tid & 3) * 8;

  auto stage = [&](int kt, int buf) {
    const unsigned kOff = (unsigned)(uintptr_t)&ldsK[buf][0] +
                          (unsigned)((tid >> 3) * (KSTR * 2) + (tid & 7) * 16);
    const unsigned vOff = (unsigned)(uintptr_t)&ldsV[buf][0] +
                          (unsigned)((tid >> 2) * (VSTR * 2) + (tid & 3) * 16);
    async_copy_b128(kOff, Kg_base + (size_t)kt * D_K);
    async_copy_b128(vOff, Vg_base + kt);
  };

  // Load Q tile (16x64) as two 16x32 A matrices, pre-scaled by 1/sqrt(64)
  H16 aq[2];
  const _Float16* Qrow = Qb + ((size_t)bh * SEQ + q0) * D_K;
#pragma unroll
  for (int c = 0; c < 2; ++c) {
    const _Float16* p = Qrow + (size_t)rr * D_K + c * 32;
    aq[c].q[0] = *(const v4i*)(p + kb);
    aq[c].q[1] = *(const v4i*)(p + kb + 8);
    aq[c].v    = aq[c].v * (_Float16)0.125f;
  }

  v8f  z[4] = {};
  float m_i[8], l_i[8];
#pragma unroll
  for (int i = 0; i < 8; ++i) { m_i[i] = -1e30f; l_i[i] = 0.0f; }

  _Float16* P = &ldsP[wave][0];

  // prologue: fill buffer 0
  stage(0, 0);
  wait_async0();
  __syncthreads();

  for (int kt = 0; kt < SEQ; kt += 32) {
    const int cur = (kt >> 5) & 1;
    if (kt + 32 < SEQ) stage(kt + 32, cur ^ 1);   // prefetch next chunk

    const _Float16* Kl = &ldsK[cur][0];
    const _Float16* Vl = &ldsV[cur][0];

    // ---- GEMM1: scores for 32 keys (two 16x16 tiles), K-dim = d_k = 64 ----
    v8f sc[2];
#pragma unroll
    for (int t = 0; t < 2; ++t) {
      const _Float16* Krow = Kl + (t * 16 + rr) * KSTR;
      H16 b0, b1;
      b0.q[0] = *(const v4i*)(Krow + ks);
      b0.q[1] = *(const v4i*)(Krow + ks + 8);
      b1.q[0] = *(const v4i*)(Krow + 32 + ks);
      b1.q[1] = *(const v4i*)(Krow + 32 + ks + 8);
      v8f s = {};
      s = wmma_f16(aq[0].v, b0.v, s);
      s = wmma_f16(aq[1].v, b1.v, s);
      const float madd =
          (1.0f - mask[(size_t)b * SEQ + kt + t * 16 + rr]) * -10000.0f;
#pragma unroll
      for (int i = 0; i < 8; ++i) s[i] += madd;
      sc[t] = s;
    }

    // ---- online softmax: row max / rescale / exp / row sum ----
#pragma unroll
    for (int i = 0; i < 8; ++i) {
      float mx = fmaxf(sc[0][i], sc[1][i]);
      mx = fmaxf(mx, __shfl_xor(mx, 1, 32));
      mx = fmaxf(mx, __shfl_xor(mx, 2, 32));
      mx = fmaxf(mx, __shfl_xor(mx, 4, 32));
      mx = fmaxf(mx, __shfl_xor(mx, 8, 32));
      const float mnew  = fmaxf(m_i[i], mx);
      const float alpha = __expf(m_i[i] - mnew);
      m_i[i] = mnew;
      l_i[i] *= alpha;
#pragma unroll
      for (int n = 0; n < 4; ++n) z[n][i] *= alpha;

      const float p0 = __expf(sc[0][i] - mnew);
      const float p1 = __expf(sc[1][i] - mnew);
      float rs = p0 + p1;
      rs += __shfl_xor(rs, 1, 32);
      rs += __shfl_xor(rs, 2, 32);
      rs += __shfl_xor(rs, 4, 32);
      rs += __shfl_xor(rs, 8, 32);
      l_i[i] += rs;

      const int r = (lane < 16) ? i : i + 8;            // C/D-layout row
      P[r * 32 + rr]      = (_Float16)p0;
      P[r * 32 + 16 + rr] = (_Float16)p1;
    }
    asm volatile("s_wait_dscnt 0" ::: "memory");        // wave-local LDS RAW

    // re-read probabilities in A layout (16x32 over key positions)
    H16 ap;
    ap.q[0] = *(const v4i*)&P[rr * 32 + kb];
    ap.q[1] = *(const v4i*)&P[rr * 32 + kb + 8];

    // ---- GEMM2: z += P(16x32) * V(32x64), V^T rows contiguous in LDS ----
#pragma unroll
    for (int n = 0; n < 4; ++n) {
      const _Float16* Vrow = Vl + (n * 16 + rr) * VSTR;
      H16 bv_;
      bv_.q[0] = *(const v4i*)(Vrow + ks);
      bv_.q[1] = *(const v4i*)(Vrow + ks + 8);
      z[n] = wmma_f16(ap.v, bv_.v, z[n]);
    }

    wait_async0();        // our prefetch done (ASYNCcnt)
    __syncthreads();      // all waves' prefetches visible / consumption done
  }

  // ---- epilogue: 1/l scaling, store Z as f16 [b, s, h*64+d] ----
  float rl[8];
#pragma unroll
  for (int i = 0; i < 8; ++i) rl[i] = 1.0f / l_i[i];
  const int h = bh & 15;
#pragma unroll
  for (int n = 0; n < 4; ++n) {
    const int d = n * 16 + rr;
#pragma unroll
    for (int i = 0; i < 8; ++i) {
      const int srow = q0 + ((lane < 16) ? i : i + 8);
      Zb[(size_t)(b * SEQ + srow) * D_MODEL + h * D_K + d] =
          (_Float16)(z[n][i] * rl[i]);
    }
  }
}

// ---------------------------------------------------------------------------
// Kernel 3: output projection.  Out = Z(f16) @ wo^T + bo  -> f32
// A tiles staged with async global->LDS (already f16); B converted in VGPRs.
// ---------------------------------------------------------------------------
__global__ __launch_bounds__(256) void out_proj_kernel(
    const _Float16* __restrict__ Zb, const float* __restrict__ W,
    const float* __restrict__ Bias, float* __restrict__ Out)
{
  __shared__ __align__(16) _Float16 ldsA[64  * LDSTR];
  __shared__ __align__(16) _Float16 ldsB[128 * LDSTR];

  const int tid  = threadIdx.x;
  const int lane = tid & 31;
  const int wave = tid >> 5;
  const int wm   = wave & 3;
  const int wn   = wave >> 2;
  const int row0 = blockIdx.y * 64;
  const int col0 = blockIdx.x * 128;

  v8f acc[4] = {};

  for (int k0 = 0; k0 < D_MODEL; k0 += 32) {
    __syncthreads();
    {   // stage A: f16 straight async copy (no VGPR round-trip)
      const int r = tid >> 2, c = (tid & 3) * 8;
      const unsigned aOff = (unsigned)(uintptr_t)&ldsA[0] +
                            (unsigned)((r * LDSTR + c) * 2);
      async_copy_b128(aOff, Zb + (size_t)(row0 + r) * D_MODEL + k0 + c);
    }
    {   // stage B^T with f32 -> f16 convert (overlaps with async A)
      const int j = tid >> 1, c = (tid & 1) * 16;
      const float* src = W + (size_t)(col0 + j) * D_MODEL + k0 + c;
      v4f w0 = *(const v4f*)src,       w1 = *(const v4f*)(src + 4);
      v4f w2 = *(const v4f*)(src + 8), w3 = *(const v4f*)(src + 12);
      H8 lo, hi;
#pragma unroll
      for (int e = 0; e < 4; ++e) {
        lo.h[e] = (_Float16)w0[e]; lo.h[4+e] = (_Float16)w1[e];
        hi.h[e] = (_Float16)w2[e]; hi.h[4+e] = (_Float16)w3[e];
      }
      *(v4i*)&ldsB[j * LDSTR + c]     = lo.q;
      *(v4i*)&ldsB[j * LDSTR + c + 8] = hi.q;
    }
    wait_async0();
    __syncthreads();

    H16 a;
    {
      const int r  = wm * 16 + (lane & 15);
      const int kbb = (lane < 16) ? 0 : 16;
      a.q[0] = *(const v4i*)&ldsA[r * LDSTR + kbb];
      a.q[1] = *(const v4i*)&ldsA[r * LDSTR + kbb + 8];
    }
#pragma unroll
    for (int nt = 0; nt < 4; ++nt) {
      const int j  = wn * 64 + nt * 16 + (lane & 15);
      const int ks = (lane < 16) ? 0 : 16;
      H16 bm;
      bm.q[0] = *(const v4i*)&ldsB[j * LDSTR + ks];
      bm.q[1] = *(const v4i*)&ldsB[j * LDSTR + ks + 8];
      acc[nt] = wmma_f16(a.v, bm.v, acc[nt]);
    }
  }

  const int rowbase = row0 + wm * 16 + ((lane < 16) ? 0 : 8);
#pragma unroll
  for (int nt = 0; nt < 4; ++nt) {
    const int j = col0 + wn * 64 + nt * 16 + (lane & 15);
    const float bias = Bias[j];
#pragma unroll
    for (int i = 0; i < 8; ++i)
      Out[(size_t)(rowbase + i) * D_MODEL + j] = acc[nt][i] + bias;
  }
}

// ---------------------------------------------------------------------------
extern "C" void kernel_launch(void* const* d_in, const int* in_sizes, int n_in,
                              void* d_out, int out_size, void* d_ws,
                              size_t ws_size, hipStream_t stream) {
  (void)in_sizes; (void)n_in; (void)out_size; (void)ws_size;
  const float* x    = (const float*)d_in[0];
  const float* y    = (const float*)d_in[1];
  const float* mask = (const float*)d_in[2];
  const float* wq   = (const float*)d_in[3];
  const float* bq   = (const float*)d_in[4];
  const float* wk   = (const float*)d_in[5];
  const float* bk   = (const float*)d_in[6];
  const float* wv   = (const float*)d_in[7];
  const float* bv   = (const float*)d_in[8];
  const float* wo   = (const float*)d_in[9];
  const float* bo   = (const float*)d_in[10];

  const size_t elems = (size_t)BATCH * SEQ * D_MODEL;   // 4,194,304
  _Float16* Qb = (_Float16*)d_ws;
  _Float16* Kb = Qb + elems;
  _Float16* Vt = Kb + elems;
  _Float16* Zb = Vt + elems;   // total 4 * elems * 2B = 32 MB of scratch

  dim3 g1(D_MODEL / 128, (BATCH * SEQ) / 64, 3);
  qkv_proj_kernel<<<g1, dim3(256), 0, stream>>>(x, y, wq, bq, wk, bk, wv, bv,
                                                Qb, Kb, Vt);

  dim3 g2(SEQ / 128, BATCH * NUM_HEADS);
  attn_kernel<<<g2, dim3(256), 0, stream>>>(Qb, Kb, Vt, mask, Zb);

  dim3 g3(D_MODEL / 128, (BATCH * SEQ) / 64);
  out_proj_kernel<<<g3, dim3(256), 0, stream>>>(Zb, wo, bo, (float*)d_out);
}
